// MotionDecoder_19189913878711
// MI455X (gfx1250) — compile-verified, matrix operand order checked by reading, not run on hardware
//
#include <hip/hip_runtime.h>
#include <hip/hip_bf16.h>

// Problem constants
#define B_   2048
#define Z_   64
#define H_   256
#define D_   8
#define T_   250
#define L_   2
#define BM   32     // batch rows per workgroup -> 64 persistent workgroups

typedef __attribute__((ext_vector_type(16))) __bf16 v16bf;
typedef __attribute__((ext_vector_type(8)))  float  v8f;

union FragA { v16bf v; unsigned u[8]; };
union FragB { v16bf v; uint4 q[2]; };
union Acc   { v8f v; float f[8]; };

// K index held by (vgpr r, lane-half h, packed pos p) for 16-bit WMMA operands
// (CDNA5 ISA 7.12.2: VGPR0 lanes0-15 K=0,1; lanes16-31 K=8,9; VGPR4-7 -> +16)
__device__ __forceinline__ int kmap(int r, int h) {
  return (r < 4 ? 2 * r : 16 + 2 * (r - 4)) + 8 * h;
}

__device__ __forceinline__ unsigned short f2bf(float f) {
  unsigned u = __float_as_uint(f);
  u += 0x7FFFu + ((u >> 16) & 1u);          // round-to-nearest-even to bf16
  return (unsigned short)(u >> 16);
}

// gfx1250 hardware transcendental tanh (TRANS32 -> co-executes with WMMA)
__device__ __forceinline__ float tanh_fast(float x) {
#if __has_builtin(__builtin_amdgcn_tanhf)
  return __builtin_amdgcn_tanhf(x);
#elif __has_builtin(__builtin_amdgcn_tanh_f32)
  return __builtin_amdgcn_tanh_f32(x);
#else
  float y;
  asm("v_tanh_f32 %0, %1" : "=v"(y) : "v"(x));
  return y;
#endif
}

// sigmoid(x) = 0.5*tanh(0.5x) + 0.5  (1 TRANS + 2 FMA, no rcp)
__device__ __forceinline__ float sigmoid_fast(float x) {
  return fmaf(tanh_fast(x * 0.5f), 0.5f, 0.5f);
}

// A fragment (16x32 bf16) gathered from an LDS row-major bf16 buffer
__device__ __forceinline__ v16bf load_a(const unsigned short* Abuf, int stride,
                                        int rowbase, int kbase, int lane) {
  FragA fa;
  const int h = lane >> 4;
  const unsigned short* rp = Abuf + (rowbase + (lane & 15)) * stride + kbase;
#pragma unroll
  for (int r = 0; r < 8; ++r)
    fa.u[r] = *(const unsigned*)(rp + kmap(r, h));   // kmap is even -> dword aligned
  return fa.v;
}

// B fragment (32x16 bf16) from pre-packed fragment-ready global buffer:
// layout dword index = (tile*32 + lane)*8 + r  -> two b128 loads per lane
__device__ __forceinline__ v16bf load_b(const unsigned* wpk, int tile, int lane) {
  FragB fb;
  const uint4* p = (const uint4*)wpk + (size_t)(tile * 32 + lane) * 2;
  fb.q[0] = p[0];
  fb.q[1] = p[1];
  return fb.v;
}

__device__ __forceinline__ v8f wmma_bf16(v16bf a, v16bf b, v8f c) {
  return __builtin_amdgcn_wmma_f32_16x16x32_bf16(
      /*neg_a=*/false, a, /*neg_b=*/false, b,
      /*c_mod=*/(short)0, c, /*reuse_a=*/false, /*reuse_b=*/false);
}

// ---------------------------------------------------------------------------
// Kernel 1: initial states  h0/c0 = (z @ W + b).reshape(L, B, H)
// reshape of [B, 512] to (2,2048,256): (l,bb,hh) <- M[l*1024 + bb/2][(bb&1)*256+hh]
// ---------------------------------------------------------------------------
__global__ __launch_bounds__(256)
void init_states_kernel(const float* __restrict__ z,
                        const float* __restrict__ Whf, const float* __restrict__ bhf,
                        const float* __restrict__ Wcf, const float* __restrict__ bcf,
                        float* __restrict__ hinit, float* __restrict__ cinit) {
  int idx = blockIdx.x * 256 + threadIdx.x;      // over L*B*H = 1048576
  int l   = idx >> 19;
  int rem = idx & ((1 << 19) - 1);
  int bb  = rem >> 8;
  int hh  = rem & 255;
  int rowb = l * 1024 + (bb >> 1);
  int col  = (bb & 1) * 256 + hh;
  float ah = bhf[col], ac = bcf[col];
  const float* zp = z + (size_t)rowb * Z_;
  for (int k = 0; k < Z_; ++k) {
    float zv = zp[k];
    ah = fmaf(zv, Whf[k * (H_ * L_) + col], ah);
    ac = fmaf(zv, Wcf[k * (H_ * L_) + col], ac);
  }
  hinit[idx] = ah;
  cinit[idx] = ac;
}

// ---------------------------------------------------------------------------
// Kernel 2: pack f32 weights into bf16 WMMA-B fragment layout.
// Tile = 32(K) x 16(N); dword (tile*32+lane)*8+r holds {bf16 W[K0][n], bf16 W[K0+1][n]}
// with n = nt*16 + (lane&15), K0 = kt*32 + kmap(r, lane>>4).
// Regions: Wih0 (K 8->32 pad), Whh0, Wih1, Whh1 (all N=1024),
//          fc_out_w (K=256, N 8->16 pad, row stride 8).
// ---------------------------------------------------------------------------
__global__ __launch_bounds__(256)
void pack_weights_kernel(const float* __restrict__ Wih0, const float* __restrict__ Whh0,
                         const float* __restrict__ Wih1, const float* __restrict__ Whh1,
                         const float* __restrict__ Wout,
                         unsigned* __restrict__ p0ih, unsigned* __restrict__ p0hh,
                         unsigned* __restrict__ p1ih, unsigned* __restrict__ p1hh,
                         unsigned* __restrict__ pout) {
  int idx = blockIdx.x * 256 + threadIdx.x;
  const int S0 = 64 * 1 * 256;   // 16384 dwords (ktiles=1)
  const int S  = 64 * 8 * 256;   // 131072 dwords (ktiles=8)
  const int S2 = 1 * 8 * 256;    // 2048 dwords (fc_out, ktiles=8, 1 n-tile)
  const float* W; unsigned* dst; int ktiles, Kact, nstride, nvalid;
  if (idx < S0)            { W = Wih0; dst = p0ih; ktiles = 1; Kact = 8;   nstride = 1024; nvalid = 1024; }
  else if (idx < S0 + S)   { idx -= S0;           W = Whh0; dst = p0hh; ktiles = 8; Kact = 256; nstride = 1024; nvalid = 1024; }
  else if (idx < S0 + 2*S) { idx -= S0 + S;       W = Wih1; dst = p1ih; ktiles = 8; Kact = 256; nstride = 1024; nvalid = 1024; }
  else if (idx < S0 + 3*S) { idx -= S0 + 2*S;     W = Whh1; dst = p1hh; ktiles = 8; Kact = 256; nstride = 1024; nvalid = 1024; }
  else                     { idx -= S0 + 3*S;     W = Wout; dst = pout; ktiles = 8; Kact = 256; nstride = 8;    nvalid = 8;    }
  if (idx >= S0 + 3*S + S2) return;   // (never true; keeps bounds explicit)
  int r     = idx & 7;
  int lanei = (idx >> 3) & 31;
  int tile  = idx >> 8;
  int nt = tile / ktiles, kt = tile % ktiles;
  int h  = lanei >> 4;
  int n  = nt * 16 + (lanei & 15);
  int K0 = kt * 32 + kmap(r, h);
  float v0 = (K0     < Kact && n < nvalid) ? W[(size_t)K0 * nstride + n]       : 0.0f;
  float v1 = (K0 + 1 < Kact && n < nvalid) ? W[(size_t)(K0 + 1) * nstride + n] : 0.0f;
  dst[(size_t)(tile * 32 + lanei) * 8 + r] =
      (unsigned)f2bf(v0) | ((unsigned)f2bf(v1) << 16);
}

// ---------------------------------------------------------------------------
// Kernel 3: persistent fused 2-layer LSTM + WMMA output projection.
// One WG = 32 batch rows, 8 wave32s, loops t = 0..249.
// LDS: double-buffered bf16 h0/h1 (GEMM A operands), f32 c0/c1 (in-place),
//      x tile (D=8 padded to K=32), fused biases.
// ---------------------------------------------------------------------------
__global__ __launch_bounds__(256, 1)
void lstm_persistent(const float* __restrict__ target,
                     const float* __restrict__ hinit,
                     const float* __restrict__ cinit,
                     const unsigned* __restrict__ wpk0ih,
                     const unsigned* __restrict__ wpk0hh,
                     const unsigned* __restrict__ wpk1ih,
                     const unsigned* __restrict__ wpk1hh,
                     const unsigned* __restrict__ wpkout,
                     const float* __restrict__ bih0, const float* __restrict__ bhh0,
                     const float* __restrict__ bih1, const float* __restrict__ bhh1,
                     const float* __restrict__ outb,
                     float* __restrict__ out) {
  extern __shared__ char smem[];
  unsigned short* h0b = (unsigned short*)smem;            // [2][BM][H]  bf16
  unsigned short* h1b = h0b + 2 * BM * H_;                // [2][BM][H]  bf16
  float* c0   = (float*)(h1b + 2 * BM * H_);              // [BM][H]
  float* c1   = c0 + BM * H_;                             // [BM][H]
  unsigned short* xb = (unsigned short*)(c1 + BM * H_);   // [BM][32] bf16 (K-padded x)
  float* bias0 = (float*)(xb + BM * 32);                  // [1024] = b_ih0+b_hh0
  float* bias1 = bias0 + 4 * H_;                          // [1024]
  float* ob    = bias1 + 4 * H_;                          // [D]

  const int tid  = threadIdx.x;
  const int b0   = blockIdx.x * BM;
  const int w    = tid >> 5;
  const int lane = tid & 31;
  const int mbase = (w & 1) * 16;      // 2 M-tiles of 16 rows
  const int ntb   = (w >> 1) * 4;      // 4 N-tiles (over H) per wave

  // --- stage constants + initial state into LDS ---
  for (int i = tid; i < 4 * H_; i += 256) {
    bias0[i] = bih0[i] + bhh0[i];
    bias1[i] = bih1[i] + bhh1[i];
  }
  if (tid < D_) ob[tid] = outb[tid];

  for (int i = tid; i < BM * H_; i += 256) {
    int row = i >> 8, col = i & (H_ - 1);
    size_t g = (size_t)(b0 + row) * H_ + col;
    c0[i] = cinit[g];
    c1[i] = cinit[(size_t)B_ * H_ + g];
    h0b[i] = f2bf(hinit[g]);
    h1b[i] = f2bf(hinit[(size_t)B_ * H_ + g]);
  }
  __syncthreads();

  int cur = 0;
  for (int t = 0; t < T_; ++t) {
    // --- stage x_t (teacher forcing: target[:, t-1], zeros at t==0) ---
    {
      int i0 = tid * 4;                       // 32*32 entries / 256 threads
#pragma unroll
      for (int e = 0; e < 4; ++e) {
        int i = i0 + e;
        int row = i >> 5, k = i & 31;
        float v = 0.0f;
        if (t > 0 && k < D_)
          v = target[((size_t)(b0 + row) * T_ + (t - 1)) * D_ + k];
        xb[i] = f2bf(v);
      }
    }
    __syncthreads();

    const unsigned short* hA0 = h0b + cur * BM * H_;
    unsigned short*       hN0 = h0b + (cur ^ 1) * BM * H_;

    // ---------------- layer 0: gates = x@Wih0 + h0@Whh0 + b ----------------
    for (int j = 0; j < 4; ++j) {
      const int ntile = ntb + j;
      Acc ai{}, af{}, ag{}, ao{};
      {  // x contribution (single padded K-tile)
        v16bf a = load_a(xb, 32, mbase, 0, lane);
        ai.v = wmma_bf16(a, load_b(wpk0ih, 0 * 16 + ntile, lane), ai.v);
        af.v = wmma_bf16(a, load_b(wpk0ih, 1 * 16 + ntile, lane), af.v);
        ag.v = wmma_bf16(a, load_b(wpk0ih, 2 * 16 + ntile, lane), ag.v);
        ao.v = wmma_bf16(a, load_b(wpk0ih, 3 * 16 + ntile, lane), ao.v);
      }
      for (int kt = 0; kt < 8; ++kt) {   // h0 contribution, K=256
        v16bf a = load_a(hA0, H_, mbase, kt * 32, lane);
        ai.v = wmma_bf16(a, load_b(wpk0hh, (0 * 16 + ntile) * 8 + kt, lane), ai.v);
        af.v = wmma_bf16(a, load_b(wpk0hh, (1 * 16 + ntile) * 8 + kt, lane), af.v);
        ag.v = wmma_bf16(a, load_b(wpk0hh, (2 * 16 + ntile) * 8 + kt, lane), ag.v);
        ao.v = wmma_bf16(a, load_b(wpk0hh, (3 * 16 + ntile) * 8 + kt, lane), ao.v);
      }
      // elementwise LSTM cell on exclusively-owned (row,col) elements
      const int col   = ntile * 16 + (lane & 15);
      const int rbase = mbase + (lane >> 4) * 8;     // C-layout: lanes>=16 -> M+8
#pragma unroll
      for (int r = 0; r < 8; ++r) {
        int idx = (rbase + r) * H_ + col;
        float iv = sigmoid_fast(ai.f[r] + bias0[col]);
        float fv = sigmoid_fast(af.f[r] + bias0[H_ + col]);
        float gv = tanh_fast(ag.f[r] + bias0[2 * H_ + col]);
        float ov = sigmoid_fast(ao.f[r] + bias0[3 * H_ + col]);
        float cn = fv * c0[idx] + iv * gv;
        c0[idx]  = cn;
        hN0[idx] = f2bf(ov * tanh_fast(cn));
      }
    }
    __syncthreads();

    const unsigned short* hA1 = h1b + cur * BM * H_;
    unsigned short*       hN1 = h1b + (cur ^ 1) * BM * H_;

    // ---------------- layer 1: gates = h0_new@Wih1 + h1@Whh1 + b ----------------
    for (int j = 0; j < 4; ++j) {
      const int ntile = ntb + j;
      Acc ai{}, af{}, ag{}, ao{};
      for (int kt = 0; kt < 8; ++kt) {
        v16bf a0 = load_a(hN0, H_, mbase, kt * 32, lane);   // layer-1 input
        v16bf a1 = load_a(hA1, H_, mbase, kt * 32, lane);   // previous h1
        ai.v = wmma_bf16(a0, load_b(wpk1ih, (0 * 16 + ntile) * 8 + kt, lane), ai.v);
        af.v = wmma_bf16(a0, load_b(wpk1ih, (1 * 16 + ntile) * 8 + kt, lane), af.v);
        ag.v = wmma_bf16(a0, load_b(wpk1ih, (2 * 16 + ntile) * 8 + kt, lane), ag.v);
        ao.v = wmma_bf16(a0, load_b(wpk1ih, (3 * 16 + ntile) * 8 + kt, lane), ao.v);
        ai.v = wmma_bf16(a1, load_b(wpk1hh, (0 * 16 + ntile) * 8 + kt, lane), ai.v);
        af.v = wmma_bf16(a1, load_b(wpk1hh, (1 * 16 + ntile) * 8 + kt, lane), af.v);
        ag.v = wmma_bf16(a1, load_b(wpk1hh, (2 * 16 + ntile) * 8 + kt, lane), ag.v);
        ao.v = wmma_bf16(a1, load_b(wpk1hh, (3 * 16 + ntile) * 8 + kt, lane), ao.v);
      }
      const int col   = ntile * 16 + (lane & 15);
      const int rbase = mbase + (lane >> 4) * 8;
#pragma unroll
      for (int r = 0; r < 8; ++r) {
        int idx = (rbase + r) * H_ + col;
        float iv = sigmoid_fast(ai.f[r] + bias1[col]);
        float fv = sigmoid_fast(af.f[r] + bias1[H_ + col]);
        float gv = tanh_fast(ag.f[r] + bias1[2 * H_ + col]);
        float ov = sigmoid_fast(ao.f[r] + bias1[3 * H_ + col]);
        float cn = fv * c1[idx] + iv * gv;
        c1[idx]  = cn;
        hN1[idx] = f2bf(ov * tanh_fast(cn));
      }
    }
    __syncthreads();

    // -------- output projection via WMMA: out[b,t,:] = h1 @ fc_out_w + b --------
    // [32 x 256] @ [256 x 8->16]: 2 M-tiles x 1 N-tile, done by waves 0 and 1.
    if (w < 2) {
      Acc acc{};
      for (int kt = 0; kt < 8; ++kt) {
        v16bf a = load_a(hN1, H_, w * 16, kt * 32, lane);
        acc.v = wmma_bf16(a, load_b(wpkout, kt, lane), acc.v);
      }
      const int d = lane & 15;
      if (d < D_) {
        const int rbase = w * 16 + (lane >> 4) * 8;
        float bb = ob[d];
#pragma unroll
        for (int r = 0; r < 8; ++r)
          out[((size_t)(b0 + rbase + r) * T_ + t) * D_ + d] = acc.f[r] + bb;
      }
    }
    __syncthreads();
    cur ^= 1;
  }
}

// ---------------------------------------------------------------------------
extern "C" void kernel_launch(void* const* d_in, const int* in_sizes, int n_in,
                              void* d_out, int out_size, void* d_ws, size_t ws_size,
                              hipStream_t stream) {
  (void)in_sizes; (void)n_in; (void)out_size; (void)ws_size;

  const float* z    = (const float*)d_in[0];
  const float* tgt  = (const float*)d_in[1];
  const float* whf  = (const float*)d_in[2];
  const float* bhf  = (const float*)d_in[3];
  const float* wcf  = (const float*)d_in[4];
  const float* bcf  = (const float*)d_in[5];
  const float* wih0 = (const float*)d_in[6];
  const float* whh0 = (const float*)d_in[7];
  const float* bih0 = (const float*)d_in[8];
  const float* bhh0 = (const float*)d_in[9];
  const float* wih1 = (const float*)d_in[10];
  const float* whh1 = (const float*)d_in[11];
  const float* bih1 = (const float*)d_in[12];
  const float* bhh1 = (const float*)d_in[13];
  const float* outw = (const float*)d_in[14];
  const float* outb = (const float*)d_in[15];

  // workspace layout
  float* hinit = (float*)d_ws;                              // L*B*H f32 (4 MB)
  float* cinit = hinit + (size_t)L_ * B_ * H_;              // 4 MB
  unsigned* p0ih = (unsigned*)(cinit + (size_t)L_ * B_ * H_);
  unsigned* p0hh = p0ih + 16384;                            // 64 KB
  unsigned* p1ih = p0hh + 131072;                           // 512 KB each
  unsigned* p1hh = p1ih + 131072;
  unsigned* pout = p1hh + 131072;                           // 8 KB

  init_states_kernel<<<(L_ * B_ * H_) / 256, 256, 0, stream>>>(
      z, whf, bhf, wcf, bcf, hinit, cinit);
  pack_weights_kernel<<<(16384 + 3 * 131072 + 2048) / 256, 256, 0, stream>>>(
      wih0, whh0, wih1, whh1, outw, p0ih, p0hh, p1ih, p1hh, pout);

  const int SMEM = 2 * BM * H_ * 2      // h0b
                 + 2 * BM * H_ * 2      // h1b
                 + 2 * BM * H_ * 4      // c0, c1
                 + BM * 32 * 2          // xb
                 + 2 * 4 * H_ * 4       // bias0, bias1
                 + D_ * 4;              // ob   -> ~141 KB (< 320 KB/WGP)
  (void)hipFuncSetAttribute((const void*)lstm_persistent,
                            hipFuncAttributeMaxDynamicSharedMemorySize, SMEM);
  lstm_persistent<<<B_ / BM, 256, SMEM, stream>>>(
      tgt, hinit, cinit, p0ih, p0hh, p1ih, p1hh, pout,
      bih0, bhh0, bih1, bhh1, outb, (float*)d_out);
}
